// GPT2_67164698575173
// MI455X (gfx1250) — matrix operand order repaired and checked by reading, not compiled, not run on hardware
//
#include <hip/hip_runtime.h>
#include <hip/hip_bf16.h>
#include <math.h>

// ---------------------------------------------------------------------------
// GPT-2 small forward for MI455X (gfx1250, wave32, WMMA 16x16x32 f16).
// GEMM stages LDS tiles in WMMA *fragment order* so fragments load as
// contiguous 32B chunks (ds_load_b128 pairs); all fragment loads for a K-step
// are issued as one batch so WMMAs overlap the in-order LDS returns.
// ---------------------------------------------------------------------------

typedef __attribute__((ext_vector_type(16))) _Float16 v16h;
typedef __attribute__((ext_vector_type(8)))  _Float16 v8h;
typedef __attribute__((ext_vector_type(8)))  float    v8f;

#define GD      768          // model dim
#define GFF     3072         // ffn dim
#define GNH     12           // heads
#define GHD     64           // head dim
#define GT      1024         // seq len
#define GB      2            // batch
#define GL      12           // layers
#define GV      50257        // vocab
#define GM      (GB * GT)    // 2048 rows

#define TM      128          // block tile M (4 waves x 32 rows)
#define TN      64           // block tile N (4 subtiles of 16)
#define TK      32           // K step (one WMMA K)

// ---------------------------------------------------------------------------
// Embedding: x[b,t,:] = tok_embed[tokens[b,t]] + pos_embed[t]
// ---------------------------------------------------------------------------
__global__ __launch_bounds__(256)
void embed_kernel(const int* __restrict__ tokens,
                  const float* __restrict__ tok_embed,
                  const float* __restrict__ pos_embed,
                  float* __restrict__ x) {
  int row = blockIdx.x;            // b*T + t
  int t   = row & (GT - 1);
  int tok = tokens[row];
  const float* te = tok_embed + (size_t)tok * GD;
  const float* pe = pos_embed + (size_t)t * GD;
  float* xr = x + (size_t)row * GD;
  for (int i = threadIdx.x; i < GD; i += 256)
    xr[i] = te[i] + pe[i];
}

// ---------------------------------------------------------------------------
// LayerNorm (block per row, D=768 = 3 elems/thread @ 256 threads), f16 output.
// ---------------------------------------------------------------------------
__global__ __launch_bounds__(256)
void layernorm_kernel(const float* __restrict__ x,
                      const float* __restrict__ w,
                      const float* __restrict__ b,
                      _Float16* __restrict__ out) {
  __shared__ float rs[256];
  __shared__ float rq[256];
  int row = blockIdx.x, tid = threadIdx.x;
  const float* xr = x + (size_t)row * GD;
  float vals[3];
  float s = 0.f, q = 0.f;
#pragma unroll
  for (int i = 0; i < 3; ++i) {
    float v = xr[tid + i * 256];
    vals[i] = v; s += v; q += v * v;
  }
  rs[tid] = s; rq[tid] = q;
  __syncthreads();
  for (int st = 128; st > 0; st >>= 1) {
    if (tid < st) { rs[tid] += rs[tid + st]; rq[tid] += rq[tid + st]; }
    __syncthreads();
  }
  float mu  = rs[0] * (1.0f / GD);
  float var = rq[0] * (1.0f / GD) - mu * mu;
  float inv = rsqrtf(var + 1e-5f);
  _Float16* orow = out + (size_t)row * GD;
#pragma unroll
  for (int i = 0; i < 3; ++i) {
    int c = tid + i * 256;
    orow[c] = (_Float16)((vals[i] - mu) * inv * w[c] + b[c]);
  }
}

// ---------------------------------------------------------------------------
// f32 -> f16 elementwise convert
// ---------------------------------------------------------------------------
__global__ __launch_bounds__(256)
void convert_kernel(const float* __restrict__ in, _Float16* __restrict__ out, int n) {
  int i = blockIdx.x * 256 + threadIdx.x;
  if (i < n) out[i] = (_Float16)in[i];
}

// ---------------------------------------------------------------------------
// WMMA GEMM: C[M,N] = A[M,K](f16) @ W[K,N](f32) + bias, tile 128x64.
//
// LDS is laid out in *fragment order* (inverse of ISA 7.12.2 tables):
//   A 16x32 subtile: (m,k) -> lane = m + 16*((k>>3)&1), half = (k&7)+8*(k>=16)
//   B 32x16 subtile: (k,n) -> lane = n + 16*((k>>3)&1), half = (k&7)+8*(k>=16)
// so each compute lane loads its whole v16h fragment as one contiguous
// 32-byte LDS read (2x ds_load_b128), no per-half packing.
//
// Block = 128 threads = 4 waves. Wave w computes rows [w*32, w*32+32) of the
// tile as 2 m-subtiles x 4 n-subtiles = 8 WMMAs per K step. All 12 fragment
// loads issue before the WMMA chain so matrix ops overlap LDS returns.
//   btrans  : W stored [N,K] (tied LM head uses tok_embed [V,D])
//   gelu    : tanh-approx GELU epilogue
//   residual: C += residual (in-place with C is safe)
//   C (f32) and/or C16 (f16) outputs, either may be null.
// ---------------------------------------------------------------------------
__global__ __launch_bounds__(128)
void gemm_wmma_kernel(const _Float16* __restrict__ A,
                      const float* __restrict__ W,
                      const float* __restrict__ bias,
                      const float* __restrict__ residual,
                      float* __restrict__ C,
                      _Float16* __restrict__ C16,
                      int M, int N, int K, int btrans, int gelu) {
  __shared__ _Float16 AsF[8][32][16];   // [m-subtile][lane][half]
  __shared__ _Float16 BsF[4][32][16];   // [n-subtile][lane][half]

  const int tid   = threadIdx.x;
  const int wave  = tid >> 5;
  const int lane  = tid & 31;
  const int l15   = lane & 15;
  const int hi    = lane >> 4;            // 0: M/K low half, 1: high half
  const int tileM = blockIdx.y * TM;
  const int tileN = blockIdx.x * TN;

  v8f acc[2][4] = {};                     // [m-sub within wave][n-sub]

  // constant staging coordinates
  const int bSub  = tid >> 5;             // B: one (subtile,lane) per thread
  const int bLn   = tid & 31;
  const int bCol  = tileN + bSub * 16 + (bLn & 15);
  const int bKoff = (bLn & 16) ? 8 : 0;   // halves 0..7 <- K = kb..kb+7, 8..15 <- K = 16+kb..16+kb+7
  const bool bOk  = (bCol < N);

  for (int k0 = 0; k0 < K; k0 += TK) {
    // ---- stage A: 256 (subtile,lane) pairs, 2 per thread ----
#pragma unroll
    for (int p = 0; p < 2; ++p) {
      int pr  = tid + p * 128;
      int sub = pr >> 5, ln = pr & 31;
      int row = tileM + sub * 16 + (ln & 15);
      int kb  = (ln & 16) ? 8 : 0;
      const _Float16* ap = A + (size_t)row * K + (k0 + kb);
      v8h a0 = *(const v8h*)ap;           // K = kb .. kb+7       -> halves 0..7
      v8h a1 = *(const v8h*)(ap + 16);    // K = 16+kb .. 16+kb+7 -> halves 8..15
      *(v8h*)&AsF[sub][ln][0] = a0;
      *(v8h*)&AsF[sub][ln][8] = a1;
    }
    // ---- stage B: one (subtile,lane) per thread, f32 -> f16 ----
    {
      v8h b0, b1;
      if (bOk) {
        if (btrans) {
          const float* wp = W + (size_t)bCol * K + (k0 + bKoff);
#pragma unroll
          for (int h = 0; h < 8; ++h) {
            b0[h] = (_Float16)wp[h];
            b1[h] = (_Float16)wp[16 + h];
          }
        } else {
          const float* wp = W + (size_t)(k0 + bKoff) * N + bCol;
#pragma unroll
          for (int h = 0; h < 8; ++h) {
            b0[h] = (_Float16)wp[(size_t)h * N];
            b1[h] = (_Float16)wp[(size_t)(16 + h) * N];
          }
        }
      } else {
#pragma unroll
        for (int h = 0; h < 8; ++h) { b0[h] = (_Float16)0.f; b1[h] = (_Float16)0.f; }
      }
      *(v8h*)&BsF[bSub][bLn][0] = b0;
      *(v8h*)&BsF[bSub][bLn][8] = b1;
    }
    // prefetch next K-slab of weights (global_prefetch_b8)
    if (k0 + TK < K && bOk) {
      const float* p = btrans ? (W + (size_t)bCol * K + (k0 + TK + bKoff))
                              : (W + (size_t)(k0 + TK + bKoff) * N + bCol);
      __builtin_prefetch(p, 0, 1);
    }
    __syncthreads();

    // ---- issue all 12 fragment loads, then the 8-WMMA chain ----
    v16h af0 = *(const v16h*)&AsF[wave * 2 + 0][lane][0];
    v16h af1 = *(const v16h*)&AsF[wave * 2 + 1][lane][0];
    v16h bf[4];
#pragma unroll
    for (int ns = 0; ns < 4; ++ns)
      bf[ns] = *(const v16h*)&BsF[ns][lane][0];
#pragma unroll
    for (int ns = 0; ns < 4; ++ns) {
      acc[0][ns] = __builtin_amdgcn_wmma_f32_16x16x32_f16(
          false, af0, false, bf[ns], (short)0, acc[0][ns], false, false);
      acc[1][ns] = __builtin_amdgcn_wmma_f32_16x16x32_f16(
          false, af1, false, bf[ns], (short)0, acc[1][ns], false, false);
    }
    __syncthreads();
  }

  // ---- epilogue per ISA 7.12.2 C/D layout (VGPR r -> M = r or r+8) ----
#pragma unroll
  for (int ms = 0; ms < 2; ++ms) {
#pragma unroll
    for (int ns = 0; ns < 4; ++ns) {
      int col = tileN + ns * 16 + l15;
      if (col >= N) continue;
      float bv = bias ? bias[col] : 0.0f;
#pragma unroll
      for (int r = 0; r < 8; ++r) {
        int row = tileM + (wave * 2 + ms) * 16 + (hi ? r + 8 : r);
        float v = acc[ms][ns][r] + bv;
        if (gelu) {
          float u = v;
          v = 0.5f * u * (1.0f + tanhf(0.7978845608f * (u + 0.044715f * u * u * u)));
        }
        size_t off = (size_t)row * N + col;
        if (residual) v += residual[off];
        if (C)   C[off]   = v;
        if (C16) C16[off] = (_Float16)v;
      }
    }
  }
}

// ---------------------------------------------------------------------------
// Attention: one wave per query row (4 rows / 128-thread block).
// qkv layout [B*T, 3D]: q at col h*64, k at 768+h*64, v at 1536+h*64.
// ---------------------------------------------------------------------------
__global__ __launch_bounds__(128)
void attention_kernel(const float* __restrict__ qkv, float* __restrict__ o) {
  __shared__ float qs[4][GHD];
  __shared__ float sc[4][GT];
  const int wave = threadIdx.x >> 5;
  const int lane = threadIdx.x & 31;
  const int t = blockIdx.x * 4 + wave;
  const int b = blockIdx.y / GNH;
  const int h = blockIdx.y % GNH;
  const size_t rowBase = (size_t)(b * GT + t) * (3 * GD);

  qs[wave][lane]      = qkv[rowBase + h * GHD + lane]      * 0.125f;
  qs[wave][lane + 32] = qkv[rowBase + h * GHD + lane + 32] * 0.125f;
  __syncthreads();

  for (int j0 = 0; j0 <= t; j0 += 32) {
    int j = j0 + lane;
    if (j <= t) {
      const float* kp = qkv + (size_t)(b * GT + j) * (3 * GD) + GD + h * GHD;
      float dot = 0.f;
#pragma unroll 8
      for (int d = 0; d < GHD; ++d) dot += qs[wave][d] * kp[d];
      sc[wave][j] = dot;
    }
  }
  float m = -1e30f;
  for (int j = lane; j <= t; j += 32) m = fmaxf(m, sc[wave][j]);
#pragma unroll
  for (int i = 16; i > 0; i >>= 1) m = fmaxf(m, __shfl_xor(m, i, 32));
  float s = 0.f;
  for (int j = lane; j <= t; j += 32) {
    float p = __expf(sc[wave][j] - m);
    sc[wave][j] = p;
    s += p;
  }
#pragma unroll
  for (int i = 16; i > 0; i >>= 1) s += __shfl_xor(s, i, 32);
  float inv = 1.0f / s;

  float a0 = 0.f, a1 = 0.f;
  for (int j = 0; j <= t; ++j) {
    float p = sc[wave][j];
    const float* vp = qkv + (size_t)(b * GT + j) * (3 * GD) + 2 * GD + h * GHD;
    a0 += p * vp[lane];
    a1 += p * vp[lane + 32];
  }
  size_t orow = (size_t)(b * GT + t) * GD + h * GHD;
  o[orow + lane]      = a0 * inv;
  o[orow + lane + 32] = a1 * inv;
}

// ---------------------------------------------------------------------------
// Orchestration
// ---------------------------------------------------------------------------
extern "C" void kernel_launch(void* const* d_in, const int* in_sizes, int n_in,
                              void* d_out, int out_size, void* d_ws, size_t ws_size,
                              hipStream_t stream) {
  (void)in_sizes; (void)n_in; (void)out_size; (void)ws_size;

  const int*   tokens    = (const int*)  d_in[0];
  const float* tok_embed = (const float*)d_in[1];
  const float* pos_embed = (const float*)d_in[2];
  const float* qkv_w     = (const float*)d_in[3];
  const float* qkv_b     = (const float*)d_in[4];
  const float* proj_w    = (const float*)d_in[5];
  const float* proj_b    = (const float*)d_in[6];
  const float* ln1_w     = (const float*)d_in[7];
  const float* ln1_b     = (const float*)d_in[8];
  const float* ln2_w     = (const float*)d_in[9];
  const float* ln2_b     = (const float*)d_in[10];
  const float* fc1_w     = (const float*)d_in[11];
  const float* fc1_b     = (const float*)d_in[12];
  const float* fc2_w     = (const float*)d_in[13];
  const float* fc2_b     = (const float*)d_in[14];
  const float* lnf_w     = (const float*)d_in[15];
  const float* lnf_b     = (const float*)d_in[16];
  float* logits = (float*)d_out;

  // scratch layout (~47 MB)
  char* wsp = (char*)d_ws;
  float*     x    = (float*)wsp;     wsp += (size_t)GM * GD * 4;       // residual stream
  _Float16*  h16  = (_Float16*)wsp;  wsp += (size_t)GM * GD * 2;       // f16 activations
  float*     qkv  = (float*)wsp;     wsp += (size_t)GM * 3 * GD * 4;   // qkv proj
  float*     attn = (float*)wsp;     wsp += (size_t)GM * GD * 4;       // attention out
  _Float16*  ff16 = (_Float16*)wsp;  wsp += (size_t)GM * GFF * 2;      // gelu(fc1) f16

  dim3 blk128(128), blk256(256);
  const int gridM = GM / TM;   // 16

  embed_kernel<<<GM, blk256, 0, stream>>>(tokens, tok_embed, pos_embed, x);

  for (int l = 0; l < GL; ++l) {
    layernorm_kernel<<<GM, blk256, 0, stream>>>(x, ln1_w + l * GD, ln1_b + l * GD, h16);
    gemm_wmma_kernel<<<dim3((3 * GD) / TN, gridM), blk128, 0, stream>>>(
        h16, qkv_w + (size_t)l * GD * 3 * GD, qkv_b + (size_t)l * 3 * GD,
        nullptr, qkv, nullptr, GM, 3 * GD, GD, 0, 0);
    attention_kernel<<<dim3(GT / 4, GB * GNH), blk128, 0, stream>>>(qkv, attn);
    convert_kernel<<<(GM * GD + 255) / 256, blk256, 0, stream>>>(attn, h16, GM * GD);
    gemm_wmma_kernel<<<dim3(GD / TN, gridM), blk128, 0, stream>>>(
        h16, proj_w + (size_t)l * GD * GD, proj_b + (size_t)l * GD,
        x, x, nullptr, GM, GD, GD, 0, 0);
    layernorm_kernel<<<GM, blk256, 0, stream>>>(x, ln2_w + l * GD, ln2_b + l * GD, h16);
    gemm_wmma_kernel<<<dim3(GFF / TN, gridM), blk128, 0, stream>>>(
        h16, fc1_w + (size_t)l * GD * GFF, fc1_b + (size_t)l * GFF,
        nullptr, nullptr, ff16, GM, GFF, GD, 0, 1);
    gemm_wmma_kernel<<<dim3(GD / TN, gridM), blk128, 0, stream>>>(
        ff16, fc2_w + (size_t)l * GFF * GD, fc2_b + (size_t)l * GD,
        x, x, nullptr, GM, GD, GFF, 0, 0);
  }

  layernorm_kernel<<<GM, blk256, 0, stream>>>(x, lnf_w, lnf_b, h16);
  gemm_wmma_kernel<<<dim3((GV + TN - 1) / TN, gridM), blk128, 0, stream>>>(
      h16, tok_embed, nullptr, nullptr, logits, nullptr, GM, GV, GD, 1, 0);
}
